// HOPE_75393855914507
// MI455X (gfx1250) — matrix operand order repaired and checked by reading, not compiled, not run on hardware
//
#include <hip/hip_runtime.h>

// ---------------------------------------------------------------------------
// HOPE forward for MI455X (gfx1250): bf16 WMMA GEMMs + register-resident
// linear-attention scan. Heavy math via v_wmma_f32_16x16x32_bf16; global->LDS
// staging via CDNA5 async-to-LDS when the toolchain exposes the builtin.
// ---------------------------------------------------------------------------

#define B_   2
#define T_   2048
#define D_   512
#define H_   8
#define V_   32000
#define L_   4
#define NL_  3
#define HID_ (4 * D_)
#define HD_  (D_ / H_)
#define M_   (B_ * T_)
#define EPS_ 1e-5f

typedef __bf16 bf16;
typedef __attribute__((ext_vector_type(16))) __bf16 v16bf;
typedef __attribute__((ext_vector_type(8)))  float  v8f;
typedef int v4i __attribute__((vector_size(16)));
typedef __attribute__((address_space(1))) v4i gv4i;  // global int4
typedef __attribute__((address_space(3))) v4i lv4i;  // LDS int4

union FragU {
  unsigned int u[8];
  v16bf v;
};

#if defined(__has_builtin)
#if __has_builtin(__builtin_amdgcn_global_load_async_to_lds_b128)
#define USE_ASYNC_LDS 1
#endif
#endif

// ---------------------------------------------------------------------------
// WMMA GEMM: C[M,N] (f32) = A[M,K](bf16, row-major) * Bt[N,K](bf16, row-major,
// i.e. B transposed).  Optional bias[N] and accumulate-into-C.
// Block tile 128(M) x 128(N), K step 32, 8 waves (256 threads); each wave owns
// a 64x32 output tile = 4x2 WMMA 16x16 accumulators.
// ---------------------------------------------------------------------------
#define BM 128
#define BN 128
#define BK 32
#define LDST 40  // padded LDS stride in bf16 elements (80B rows, 16B aligned)

__global__ __launch_bounds__(256) void k_gemm_bf16(
    const bf16* __restrict__ A, const bf16* __restrict__ Bt,
    float* __restrict__ C, const float* __restrict__ bias,
    int Mdim, int Ndim, int Kdim, int accFlag) {
  __shared__ bf16 As[BM * LDST];
  __shared__ bf16 Bs[BN * LDST];

  const int tid   = threadIdx.x;
  const int lane  = tid & 31;
  const int wave  = tid >> 5;
  const int waveM = wave >> 2;  // 0..1 -> 64 rows
  const int waveN = wave & 3;   // 0..3 -> 32 cols
  const int mBase = blockIdx.y * BM;
  const int nBase = blockIdx.x * BN;

  v8f acc[4][2];
#pragma unroll
  for (int i = 0; i < 4; ++i)
#pragma unroll
    for (int j = 0; j < 2; ++j)
#pragma unroll
      for (int e = 0; e < 8; ++e) acc[i][j][e] = 0.0f;

  const int r0 = tid >> 2;  // 0..63 (also handles row r0+64)
  const int c0 = tid & 3;   // 8-bf16 chunk

  for (int k0 = 0; k0 < Kdim; k0 += BK) {
    // ---- global -> LDS staging (128-bit, two rows per thread per matrix) ----
#ifdef USE_ASYNC_LDS
    {
      auto cp = [](bf16* ldsDst, const bf16* gSrc) {
        __builtin_amdgcn_global_load_async_to_lds_b128(
            (gv4i*)gSrc, (lv4i*)ldsDst, 0, 0);
      };
      cp(As + r0 * LDST + c0 * 8,        A + (size_t)(mBase + r0) * Kdim + k0 + c0 * 8);
      cp(As + (r0 + 64) * LDST + c0 * 8, A + (size_t)(mBase + r0 + 64) * Kdim + k0 + c0 * 8);
      cp(Bs + r0 * LDST + c0 * 8,        Bt + (size_t)(nBase + r0) * Kdim + k0 + c0 * 8);
      cp(Bs + (r0 + 64) * LDST + c0 * 8, Bt + (size_t)(nBase + r0 + 64) * Kdim + k0 + c0 * 8);
    }
#if __has_builtin(__builtin_amdgcn_s_wait_asynccnt)
    __builtin_amdgcn_s_wait_asynccnt(0);
#else
    asm volatile("s_wait_asynccnt 0x0" ::: "memory");
#endif
#else
    *(uint4*)(As + r0 * LDST + c0 * 8) =
        *(const uint4*)(A + (size_t)(mBase + r0) * Kdim + k0 + c0 * 8);
    *(uint4*)(As + (r0 + 64) * LDST + c0 * 8) =
        *(const uint4*)(A + (size_t)(mBase + r0 + 64) * Kdim + k0 + c0 * 8);
    *(uint4*)(Bs + r0 * LDST + c0 * 8) =
        *(const uint4*)(Bt + (size_t)(nBase + r0) * Kdim + k0 + c0 * 8);
    *(uint4*)(Bs + (r0 + 64) * LDST + c0 * 8) =
        *(const uint4*)(Bt + (size_t)(nBase + r0 + 64) * Kdim + k0 + c0 * 8);
#endif
    __syncthreads();

    // ---- LDS -> fragments per CDNA5 16-bit WMMA lane layout ----
    FragU afr[4], bfr[2];
    const int mloc  = lane & 15;
    const int khalf = (lane >> 4) & 1;
#pragma unroll
    for (int mt = 0; mt < 4; ++mt) {
      const int row = waveM * 64 + mt * 16 + mloc;
#pragma unroll
      for (int vv = 0; vv < 8; ++vv) {
        const int kl = ((vv < 4) ? 2 * vv : 16 + 2 * (vv - 4)) + 8 * khalf;
        afr[mt].u[vv] = *(const unsigned int*)(As + row * LDST + kl);
      }
    }
#pragma unroll
    for (int nt = 0; nt < 2; ++nt) {
      const int rowb = waveN * 32 + nt * 16 + mloc;
#pragma unroll
      for (int vv = 0; vv < 8; ++vv) {
        const int kl = ((vv < 4) ? 2 * vv : 16 + 2 * (vv - 4)) + 8 * khalf;
        bfr[nt].u[vv] = *(const unsigned int*)(Bs + rowb * LDST + kl);
      }
    }

#pragma unroll
    for (int mt = 0; mt < 4; ++mt)
#pragma unroll
      for (int nt = 0; nt < 2; ++nt)
        acc[mt][nt] = __builtin_amdgcn_wmma_f32_16x16x32_bf16(
            false, afr[mt].v, false, bfr[nt].v, (short)0, acc[mt][nt],
            false, false);
    __syncthreads();
  }

  // ---- epilogue: bias add, optional accumulate ----
#pragma unroll
  for (int mt = 0; mt < 4; ++mt) {
#pragma unroll
    for (int nt = 0; nt < 2; ++nt) {
      const int n = nBase + waveN * 32 + nt * 16 + (lane & 15);
      const float bval = bias ? bias[n] : 0.0f;
#pragma unroll
      for (int vv = 0; vv < 8; ++vv) {
        const int m = mBase + waveM * 64 + mt * 16 + vv + 8 * ((lane >> 4) & 1);
        const size_t idx = (size_t)m * Ndim + n;
        const float val = acc[mt][nt][vv] + bval;
        if (accFlag) C[idx] += val; else C[idx] = val;
      }
    }
  }
}

// ---------------------------------------------------------------------------
// Linear-attention scan: one workgroup per head; mem[64x64] in registers
// (16 f32 per thread).  Sequential over T, batch handled inline.
// ---------------------------------------------------------------------------
__global__ __launch_bounds__(256) void k_attn(const float* __restrict__ q,
                                              const float* __restrict__ k,
                                              const float* __restrict__ v,
                                              float* __restrict__ y) {
  const int h   = blockIdx.x;
  const int tid = threadIdx.x;
  const int d   = tid >> 2;         // mem row owned
  const int kb  = (tid & 3) << 4;   // mem col base (16 cols per thread)

  float mem[16];
#pragma unroll
  for (int j = 0; j < 16; ++j) mem[j] = 0.0f;

  __shared__ float qs[2][64], ks[2][64], vs[2][64];

  for (int t = 0; t < T_; ++t) {
    if (tid < 128) {
      const int b = tid >> 6, j = tid & 63;
      const size_t base = ((size_t)(b * T_ + t)) * D_ + h * HD_ + j;
      qs[b][j] = q[base];
      ks[b][j] = k[base];
      vs[b][j] = v[base];
    }
    __syncthreads();

    const float vd0 = vs[0][d], vd1 = vs[1][d];
    float p0 = 0.0f, p1 = 0.0f;
#pragma unroll
    for (int j = 0; j < 16; ++j) {
      mem[j] += 0.5f * (vd0 * ks[0][kb + j] + vd1 * ks[1][kb + j]);
      p0 += mem[j] * qs[0][kb + j];
      p1 += mem[j] * qs[1][kb + j];
    }
    // reduce over the 4 lanes that own row d (same wave32)
    p0 += __shfl_xor(p0, 1); p0 += __shfl_xor(p0, 2);
    p1 += __shfl_xor(p1, 1); p1 += __shfl_xor(p1, 2);
    if ((tid & 3) == 0) {
      y[((size_t)(0 * T_ + t)) * D_ + h * HD_ + d] = p0;
      y[((size_t)(1 * T_ + t)) * D_ + h * HD_ + d] = p1;
    }
    __syncthreads();
  }
}

// ---------------------------------------------------------------------------
// Elementwise / setup kernels
// ---------------------------------------------------------------------------
__global__ void k_embed(const float* __restrict__ emb, const float* __restrict__ pos,
                        const int* __restrict__ ids, float* __restrict__ x) {
  const size_t idx = (size_t)blockIdx.x * blockDim.x + threadIdx.x;
  if (idx >= (size_t)M_ * D_) return;
  const int row = (int)(idx / D_), col = (int)(idx % D_);
  const int t = row % T_;
  x[idx] = emb[(size_t)ids[row] * D_ + col] + pos[(size_t)t * D_ + col];
}

__global__ __launch_bounds__(256) void k_layernorm(
    const float* __restrict__ x, const float* __restrict__ g,
    const float* __restrict__ b, bf16* __restrict__ out) {
  const int row = blockIdx.x, tid = threadIdx.x;
  const float* xr = x + (size_t)row * D_;
  const float a0 = xr[tid], a1 = xr[tid + 256];
  __shared__ float r1[256], r2[256];
  r1[tid] = a0 + a1;
  r2[tid] = a0 * a0 + a1 * a1;
  __syncthreads();
  for (int off = 128; off > 0; off >>= 1) {
    if (tid < off) { r1[tid] += r1[tid + off]; r2[tid] += r2[tid + off]; }
    __syncthreads();
  }
  const float mu   = r1[0] * (1.0f / D_);
  const float var  = r2[0] * (1.0f / D_) - mu * mu;
  const float rstd = rsqrtf(var + EPS_);
  out[(size_t)row * D_ + tid]       = (bf16)((a0 - mu) * rstd * g[tid] + b[tid]);
  out[(size_t)row * D_ + tid + 256] = (bf16)((a1 - mu) * rstd * g[tid + 256] + b[tid + 256]);
}

// W[K,N] f32  ->  Wt[N,K] bf16
__global__ void k_transpose_bf16(const float* __restrict__ in, bf16* __restrict__ out,
                                 int K, int N) {
  const size_t idx = (size_t)blockIdx.x * blockDim.x + threadIdx.x;
  if (idx >= (size_t)K * N) return;
  const int n = (int)(idx / K), k = (int)(idx % K);
  out[idx] = (bf16)in[(size_t)k * N + n];
}

__global__ void k_convert_bf16(const float* __restrict__ in, bf16* __restrict__ out,
                               size_t total) {
  const size_t idx = (size_t)blockIdx.x * blockDim.x + threadIdx.x;
  if (idx < total) out[idx] = (bf16)in[idx];
}

__global__ void k_gelu_bias(const float* __restrict__ h, const float* __restrict__ bias,
                            bf16* __restrict__ out, int Ndim, size_t total) {
  const size_t idx = (size_t)blockIdx.x * blockDim.x + threadIdx.x;
  if (idx >= total) return;
  const int col = (int)(idx % Ndim);
  const float t = h[idx] + bias[col];
  const float u = t + 0.044715f * t * t * t;
  out[idx] = (bf16)(0.5f * t * (1.0f + tanhf(0.7978845608028654f * u)));
}

__global__ void k_resid(float* __restrict__ x, const float* __restrict__ y,
                        float scale, size_t total) {
  const size_t idx = (size_t)blockIdx.x * blockDim.x + threadIdx.x;
  if (idx < total) x[idx] += y[idx] * scale;
}

// ---------------------------------------------------------------------------
// Orchestration
// ---------------------------------------------------------------------------
extern "C" void kernel_launch(void* const* d_in, const int* in_sizes, int n_in,
                              void* d_out, int out_size, void* d_ws, size_t ws_size,
                              hipStream_t stream) {
  const float* emb   = (const float*)d_in[0];
  const float* pos   = (const float*)d_in[1];
  const float* ln1_g = (const float*)d_in[2];
  const float* ln1_b = (const float*)d_in[3];
  const float* Wq    = (const float*)d_in[4];
  const float* Wk    = (const float*)d_in[5];
  const float* Wv    = (const float*)d_in[6];
  const float* Wo    = (const float*)d_in[7];
  const float* ln2_g = (const float*)d_in[8];
  const float* ln2_b = (const float*)d_in[9];
  const float* W1    = (const float*)d_in[10];
  const float* b1    = (const float*)d_in[11];
  const float* W2    = (const float*)d_in[12];
  const float* b2    = (const float*)d_in[13];
  const float* lnf_g = (const float*)d_in[14];
  const float* lnf_b = (const float*)d_in[15];
  const int*   ids   = (const int*)d_in[16];
  float* out = (float*)d_out;

  char* ws = (char*)d_ws;
  size_t off = 0;
  auto alloc = [&](size_t bytes) -> void* {
    void* p = ws + off;
    off += (bytes + 255) & ~(size_t)255;
    return p;
  };

  float* x     = (float*)alloc((size_t)M_ * D_ * 4);
  bf16*  xn    = (bf16*) alloc((size_t)M_ * D_ * 2);
  float* qf    = (float*)alloc((size_t)M_ * D_ * 4);
  float* kf    = (float*)alloc((size_t)M_ * D_ * 4);
  float* vf    = (float*)alloc((size_t)M_ * D_ * 4);
  float* ay    = (float*)alloc((size_t)M_ * D_ * 4);
  bf16*  aybf  = (bf16*) alloc((size_t)M_ * D_ * 2);
  float* hf    = (float*)alloc((size_t)M_ * HID_ * 4);
  bf16*  hbf   = (bf16*) alloc((size_t)M_ * HID_ * 2);
  float* yf    = (float*)alloc((size_t)M_ * D_ * 4);
  bf16*  wt    = (bf16*) alloc((size_t)HID_ * D_ * 2);
  bf16*  embbf = (bf16*) alloc((size_t)V_ * D_ * 2);

  auto gemm = [&](const bf16* A, const bf16* Bt, float* C, const float* bias,
                  int Mdim, int Ndim, int Kdim, int accFlag) {
    dim3 grid(Ndim / BN, Mdim / BM);
    k_gemm_bf16<<<grid, 256, 0, stream>>>(A, Bt, C, bias, Mdim, Ndim, Kdim, accFlag);
  };
  auto transpose = [&](const float* in, bf16* o, int K, int N) {
    size_t tot = (size_t)K * N;
    k_transpose_bf16<<<(unsigned)((tot + 255) / 256), 256, 0, stream>>>(in, o, K, N);
  };

  // setup: emb -> bf16 (serves as Bt for tied logits head), token embed + pos
  {
    size_t tot = (size_t)V_ * D_;
    k_convert_bf16<<<(unsigned)((tot + 255) / 256), 256, 0, stream>>>(emb, embbf, tot);
  }
  {
    size_t tot = (size_t)M_ * D_;
    k_embed<<<(unsigned)((tot + 255) / 256), 256, 0, stream>>>(emb, pos, ids, x);
  }

  for (int l = 0; l < L_; ++l) {
    // ---- self-modulating attention block ----
    k_layernorm<<<M_, 256, 0, stream>>>(x, ln1_g + (size_t)l * D_, ln1_b + (size_t)l * D_, xn);

    const float* Wproj[3] = {Wq + (size_t)l * D_ * D_, Wk + (size_t)l * D_ * D_,
                             Wv + (size_t)l * D_ * D_};
    float* qkv[3] = {qf, kf, vf};
    for (int p = 0; p < 3; ++p) {
      transpose(Wproj[p], wt, D_, D_);
      gemm(xn, wt, qkv[p], nullptr, M_, D_, D_, 0);
    }

    k_attn<<<H_, 256, 0, stream>>>(qf, kf, vf, ay);
    {
      size_t tot = (size_t)M_ * D_;
      k_convert_bf16<<<(unsigned)((tot + 255) / 256), 256, 0, stream>>>(ay, aybf, tot);
    }
    transpose(Wo + (size_t)l * D_ * D_, wt, D_, D_);
    gemm(aybf, wt, x, nullptr, M_, D_, D_, /*acc=*/1);  // residual: x += attn @ Wo

    // ---- continuum memory system (3 parallel MLPs, averaged) ----
    k_layernorm<<<M_, 256, 0, stream>>>(x, ln2_g + (size_t)l * D_, ln2_b + (size_t)l * D_, xn);
    for (int n = 0; n < NL_; ++n) {
      const float* W1p = W1 + ((size_t)l * NL_ + n) * D_ * HID_;
      const float* W2p = W2 + ((size_t)l * NL_ + n) * HID_ * D_;
      const float* b1p = b1 + ((size_t)l * NL_ + n) * HID_;
      const float* b2p = b2 + ((size_t)l * NL_ + n) * D_;
      transpose(W1p, wt, D_, HID_);
      gemm(xn, wt, hf, nullptr, M_, HID_, D_, 0);
      {
        size_t tot = (size_t)M_ * HID_;
        k_gelu_bias<<<(unsigned)((tot + 255) / 256), 256, 0, stream>>>(hf, b1p, hbf, HID_, tot);
      }
      transpose(W2p, wt, HID_, D_);
      gemm(hbf, wt, yf, b2p, M_, D_, HID_, /*acc=*/(n > 0));
    }
    {
      size_t tot = (size_t)M_ * D_;
      k_resid<<<(unsigned)((tot + 255) / 256), 256, 0, stream>>>(x, yf, 1.0f / NL_, tot);
    }
  }

  // ---- final LN + tied-embedding logits (NT GEMM: Bt == emb row-major) ----
  k_layernorm<<<M_, 256, 0, stream>>>(x, lnf_g, lnf_b, xn);
  gemm(xn, embbf, out, nullptr, M_, V_, D_, 0);
}
